// MetaBaseline_34428457844826
// MI455X (gfx1250) — matrix operand order repaired and checked by reading, not compiled, not run on hardware
//
#include <hip/hip_runtime.h>
#include <hip/hip_bf16.h>
#include <math.h>

// ---------------------------------------------------------------------------
// Problem constants (from setup_inputs)
// ---------------------------------------------------------------------------
#define Bc   4
#define WAY  5
#define SHOT 5
#define Cc   640
#define HW   100
#define Qc   50
#define SHW      500            // shot*hw support descriptors per (b,way)
#define SHW_PAD  512            // padded to 32 column tiles of 16
#define QR       100            // query descriptors per (b,q)
#define QR_PAD   128            // padded to 4 row-pair tiles of 32
#define TILESM   4              // 32-row slabs per query

typedef _Float16 v16h __attribute__((ext_vector_type(16)));
typedef float    v8f  __attribute__((ext_vector_type(8)));

union HalfVec { uint4 u[2]; v16h v; };

__device__ __forceinline__ float wave_sum32(float v) {
#pragma unroll
  for (int off = 16; off >= 1; off >>= 1) v += __shfl_down(v, off, 32);
  return v;  // valid in lane 0
}

// sorted ascending top-5: t[0] is the smallest kept value
__device__ __forceinline__ void top5_insert(float (&t)[5], float v) {
  if (v > t[0]) {
    t[0] = v;
#pragma unroll
    for (int i = 0; i < 4; ++i) {
      if (t[i] > t[i + 1]) { float tmp = t[i]; t[i] = t[i + 1]; t[i + 1] = tmp; }
    }
  }
}

// ---------------------------------------------------------------------------
// init: zero accum, pass-through scalars
// ---------------------------------------------------------------------------
__global__ void k_init(float* accum, const float* rcos, const float* rdn4, float* out) {
  int i = blockIdx.x * blockDim.x + threadIdx.x;
  if (i < Bc * Qc * WAY) accum[i] = 0.0f;
  if (i == 0) { out[2 * Bc * Qc * WAY + 0] = rcos[0]; out[2 * Bc * Qc * WAY + 1] = rdn4[0]; }
}

// ---------------------------------------------------------------------------
// support descriptors: gather channels (stride hw), L2-normalize, emit f16
// ---------------------------------------------------------------------------
__global__ __launch_bounds__(32) void k_prep_support(const float* __restrict__ xs,
                                                     _Float16* __restrict__ sdesc) {
  int blk = blockIdx.x;                 // (b*WAY + w)*SHW_PAD + y
  int y  = blk % SHW_PAD;
  int bw = blk / SHW_PAD;
  int lane = threadIdx.x;
  _Float16* dst = sdesc + (size_t)blk * Cc;
  if (y >= SHW) {
    for (int c = lane; c < Cc; c += 32) dst[c] = (_Float16)0.0f;
    return;
  }
  int s = y / HW, pos = y % HW;
  const float* src = xs + ((size_t)bw * SHOT + s) * Cc * HW + pos;
  float vals[20]; float ssq = 0.0f;
#pragma unroll
  for (int i = 0; i < 20; ++i) {
    int c = lane + i * 32;
    float v = src[(size_t)c * HW];
    vals[i] = v; ssq += v * v;
  }
  ssq = wave_sum32(ssq);
  ssq = __shfl(ssq, 0, 32);
  float inv = rsqrtf(fmaxf(ssq, 1e-12f));
#pragma unroll
  for (int i = 0; i < 20; ++i) dst[lane + i * 32] = (_Float16)(vals[i] * inv);
}

// ---------------------------------------------------------------------------
// query descriptors: same, padded to QR_PAD rows per query
// ---------------------------------------------------------------------------
__global__ __launch_bounds__(32) void k_prep_query(const float* __restrict__ xq,
                                                   _Float16* __restrict__ qdesc) {
  int blk = blockIdx.x;                 // (b*Q + q)*QR_PAD + pos
  int pos = blk % QR_PAD;
  int bq  = blk / QR_PAD;
  int lane = threadIdx.x;
  _Float16* dst = qdesc + (size_t)blk * Cc;
  if (pos >= QR) {
    for (int c = lane; c < Cc; c += 32) dst[c] = (_Float16)0.0f;
    return;
  }
  const float* src = xq + (size_t)bq * Cc * HW + pos;
  float vals[20]; float ssq = 0.0f;
#pragma unroll
  for (int i = 0; i < 20; ++i) {
    int c = lane + i * 32;
    float v = src[(size_t)c * HW];
    vals[i] = v; ssq += v * v;
  }
  ssq = wave_sum32(ssq);
  ssq = __shfl(ssq, 0, 32);
  float inv = rsqrtf(fmaxf(ssq, 1e-12f));
#pragma unroll
  for (int i = 0; i < 20; ++i) dst[lane + i * 32] = (_Float16)(vals[i] * inv);
}

// ---------------------------------------------------------------------------
// proto[b][w][c] = mean over (shot,pos)
// ---------------------------------------------------------------------------
__global__ void k_proto(const float* __restrict__ xs, float* __restrict__ proto) {
  int tid = blockIdx.x * blockDim.x + threadIdx.x;   // Bc*WAY*Cc = 12800
  if (tid >= Bc * WAY * Cc) return;
  int c = tid % Cc;
  int bw = tid / Cc;
  float s = 0.0f;
  for (int sh = 0; sh < SHOT; ++sh) {
    const float* p = xs + (((size_t)bw * SHOT + sh) * Cc + c) * HW;
    for (int pos = 0; pos < HW; ++pos) s += p[pos];
  }
  proto[tid] = s * (1.0f / (SHOT * HW));
}

// ---------------------------------------------------------------------------
// q_mean[b][q][c] = mean over pos
// ---------------------------------------------------------------------------
__global__ void k_qmean(const float* __restrict__ xq, float* __restrict__ qmean) {
  int tid = blockIdx.x * blockDim.x + threadIdx.x;   // Bc*Qc*Cc = 128000
  if (tid >= Bc * Qc * Cc) return;
  const float* p = xq + (size_t)tid * HW;
  float s = 0.0f;
  for (int pos = 0; pos < HW; ++pos) s += p[pos];
  qmean[tid] = s * (1.0f / HW);
}

// ---------------------------------------------------------------------------
// cosine logits: one wave per (b,q,w)
// ---------------------------------------------------------------------------
__global__ __launch_bounds__(32) void k_cos(const float* __restrict__ qmean,
                                            const float* __restrict__ proto,
                                            float* __restrict__ out) {
  int blk = blockIdx.x;                 // (b*Q + q)*WAY + w
  int w  = blk % WAY;
  int bq = blk / WAY;
  int b  = bq / Qc;
  int lane = threadIdx.x;
  const float* qp = qmean + (size_t)bq * Cc;
  const float* pp = proto + ((size_t)b * WAY + w) * Cc;
  float dot = 0.0f, nq = 0.0f, np = 0.0f;
#pragma unroll
  for (int i = 0; i < 20; ++i) {
    int c = lane + i * 32;
    float qv = qp[c], pv = pp[c];
    dot += qv * pv; nq += qv * qv; np += pv * pv;
  }
  dot = wave_sum32(dot);
  nq  = wave_sum32(nq);
  np  = wave_sum32(np);
  if (lane == 0) out[blk] = dot * rsqrtf(fmaxf(nq * np, 1e-24f));
}

// ---------------------------------------------------------------------------
// Fused WMMA GEMM + streaming top-5, 2x2 register blocking:
//   one wave per (b, q, 32-row slab); loop over 5 ways x 16 32-column passes.
//   A (32x640 f16) staged in LDS; B streamed from global; 4 f32 accumulators
//   (2 row slabs x 2 col tiles) -> each A and B fragment feeds 2 WMMAs.
//   Each lane owns one of the 32 rows for the top-5 scan (no lane merging).
// ---------------------------------------------------------------------------
__global__ __launch_bounds__(32) void k_dn4(const _Float16* __restrict__ qdesc,
                                            const _Float16* __restrict__ sdesc,
                                            float* __restrict__ accum) {
  __shared__ __align__(16) _Float16 alds[32 * Cc];   // 40 KB
  __shared__ __align__(16) float ctile[32][36];      // 32x32 stage, padded stride

  int blk = blockIdx.x;                 // (b*Q + q)*TILESM + tm
  int tm = blk % TILESM;
  int bq = blk / TILESM;
  int b  = bq / Qc;

  int lane = threadIdx.x;
  int col  = lane & 15;                 // N within a 16-col tile / M-row index
  int hi   = lane >> 4;
  int kg8  = hi * 8;

  // ---- stage A slab (32 rows x 640 halves) into LDS: one row per lane ----
  {
    const _Float16* src = qdesc + ((size_t)bq * QR_PAD + tm * 32 + lane) * Cc;
    _Float16* dst = alds + lane * Cc;
    const uint4* s4 = (const uint4*)src;
    uint4* d4 = (uint4*)dst;
#pragma unroll
    for (int i = 0; i < Cc * 2 / 16; ++i) d4[i] = s4[i];   // 80 x 16B
  }
  __syncthreads();

  const _Float16* arow0 = alds + col * Cc;          // slab 0, row = lane&15
  const _Float16* arow1 = alds + (16 + col) * Cc;   // slab 1

  for (int w = 0; w < WAY; ++w) {
    float t[5];
#pragma unroll
    for (int i = 0; i < 5; ++i) t[i] = -2.0f;       // sims are in [-1,1]

    const _Float16* sbase = sdesc + ((size_t)(b * WAY + w)) * SHW_PAD * Cc;

    for (int cp = 0; cp < SHW_PAD / 32; ++cp) {     // 16 passes of 32 columns
      v8f a00 = {}, a01 = {}, a10 = {}, a11 = {};   // [slab][ntile]
      const _Float16* brow0 = sbase + (size_t)(cp * 32 + col) * Cc + hi * 8;
      const _Float16* brow1 = sbase + (size_t)(cp * 32 + 16 + col) * Cc + hi * 8;

#pragma unroll 4
      for (int k0 = 0; k0 < Cc; k0 += 32) {
        HalfVec A0, A1, B0, B1;
        A0.u[0] = *(const uint4*)(arow0 + k0 + kg8);
        A0.u[1] = *(const uint4*)(arow0 + k0 + 16 + kg8);
        A1.u[0] = *(const uint4*)(arow1 + k0 + kg8);
        A1.u[1] = *(const uint4*)(arow1 + k0 + 16 + kg8);
        B0.u[0] = *(const uint4*)(brow0 + k0);
        B0.u[1] = *(const uint4*)(brow0 + k0 + 16);
        B1.u[0] = *(const uint4*)(brow1 + k0);
        B1.u[1] = *(const uint4*)(brow1 + k0 + 16);
        a00 = __builtin_amdgcn_wmma_f32_16x16x32_f16(false, A0.v, false, B0.v, (short)0, a00, false, false);
        a01 = __builtin_amdgcn_wmma_f32_16x16x32_f16(false, A0.v, false, B1.v, (short)0, a01, false, false);
        a10 = __builtin_amdgcn_wmma_f32_16x16x32_f16(false, A1.v, false, B0.v, (short)0, a10, false, false);
        a11 = __builtin_amdgcn_wmma_f32_16x16x32_f16(false, A1.v, false, B1.v, (short)0, a11, false, false);
      }

      // scatter the four 16x16 C tiles: VGPR i holds M = hi*8+i, N = lane&15
      __syncthreads();
#pragma unroll
      for (int i = 0; i < 8; ++i) {
        ctile[hi * 8 + i][col]           = a00[i];
        ctile[hi * 8 + i][16 + col]      = a01[i];
        ctile[16 + hi * 8 + i][col]      = a10[i];
        ctile[16 + hi * 8 + i][16 + col] = a11[i];
      }
      __syncthreads();

      // top-5 scan: lane owns row `lane`, reads 32 columns as float4
      int ybase = cp * 32;
      const float4* crow = (const float4*)(&ctile[lane][0]);
#pragma unroll
      for (int j = 0; j < 8; ++j) {
        float4 v4 = crow[j];
        int y = ybase + j * 4;
        if (y + 0 < SHW) top5_insert(t, v4.x);
        if (y + 1 < SHW) top5_insert(t, v4.y);
        if (y + 2 < SHW) top5_insert(t, v4.z);
        if (y + 3 < SHW) top5_insert(t, v4.w);
      }
    }

    // pad rows (pos >= 100) have all-zero sims -> ssq contribution 0 naturally
    float ssq = 0.0f;
#pragma unroll
    for (int i = 0; i < 5; ++i) ssq += t[i] * t[i];
    ssq = wave_sum32(ssq);
    if (lane == 0) atomicAdd(&accum[bq * WAY + w], ssq);
  }
}

// ---------------------------------------------------------------------------
// finalize: logits_dn4 = sqrt(sum_of_squares) / (neighbor_k * Q)
// ---------------------------------------------------------------------------
__global__ void k_finalize(const float* __restrict__ accum, const int* __restrict__ nk,
                           float* __restrict__ out) {
  int i = blockIdx.x * blockDim.x + threadIdx.x;
  if (i < Bc * Qc * WAY) {
    float denom = (float)(nk[0] * Qc);
    out[Bc * Qc * WAY + i] = sqrtf(fmaxf(accum[i], 0.0f)) / denom;
  }
}

// ---------------------------------------------------------------------------
// launch
// ---------------------------------------------------------------------------
extern "C" void kernel_launch(void* const* d_in, const int* in_sizes, int n_in,
                              void* d_out, int out_size, void* d_ws, size_t ws_size,
                              hipStream_t stream) {
  const float* xs   = (const float*)d_in[0];   // [4,5,5,640,10,10]
  const float* xq   = (const float*)d_in[1];   // [4,50,640,10,10]
  const float* rcos = (const float*)d_in[2];
  const float* rdn4 = (const float*)d_in[3];
  const int*   nk   = (const int*)d_in[4];
  float* out = (float*)d_out;

  // workspace layout (all 16B-aligned offsets)
  const size_t sdesc_bytes = (size_t)Bc * WAY * SHW_PAD * Cc * sizeof(_Float16); // 13,107,200
  const size_t qdesc_bytes = (size_t)Bc * Qc * QR_PAD * Cc * sizeof(_Float16);   // 32,768,000
  char* ws = (char*)d_ws;
  _Float16* sdesc = (_Float16*)ws;
  _Float16* qdesc = (_Float16*)(ws + sdesc_bytes);
  float* proto = (float*)(ws + sdesc_bytes + qdesc_bytes);
  float* qmean = proto + Bc * WAY * Cc;
  float* accum = qmean + Bc * Qc * Cc;

  k_init<<<4, 256, 0, stream>>>(accum, rcos, rdn4, out);
  k_prep_support<<<Bc * WAY * SHW_PAD, 32, 0, stream>>>(xs, sdesc);
  k_prep_query<<<Bc * Qc * QR_PAD, 32, 0, stream>>>(xq, qdesc);
  k_proto<<<(Bc * WAY * Cc + 255) / 256, 256, 0, stream>>>(xs, proto);
  k_qmean<<<(Bc * Qc * Cc + 255) / 256, 256, 0, stream>>>(xq, qmean);
  k_cos<<<Bc * Qc * WAY, 32, 0, stream>>>(qmean, proto, out);
  k_dn4<<<Bc * Qc * TILESM, 32, 0, stream>>>(qdesc, sdesc, accum);
  k_finalize<<<(Bc * Qc * WAY + 255) / 256, 256, 0, stream>>>(accum, nk, out);
}
